// Seq2SeqLstm_83476984365338
// MI455X (gfx1250) — matrix operand order repaired and checked by reading, not compiled, not run on hardware
//
#include <hip/hip_runtime.h>

// ---------------------------------------------------------------------------
// Types / helpers
// ---------------------------------------------------------------------------
typedef __bf16 bf16_t;
typedef __attribute__((ext_vector_type(16))) __bf16 bf16x16;
typedef __attribute__((ext_vector_type(8)))  __bf16 bf16x8;
typedef __attribute__((ext_vector_type(8)))  float  f32x8;
typedef __attribute__((ext_vector_type(4)))  unsigned int u32x4;
typedef __attribute__((ext_vector_type(8)))  int i32x8;
typedef __attribute__((ext_vector_type(4)))  int i32x4;

#if defined(__has_builtin)
#if __has_builtin(__builtin_amdgcn_tensor_load_to_lds) && __has_builtin(__builtin_amdgcn_s_wait_tensorcnt)
#define USE_TDM 1
#endif
#endif
#ifndef USE_TDM
#define USE_TDM 0
#endif

__device__ __forceinline__ int lane_id() { return (int)(threadIdx.x & 31u); }

__device__ __forceinline__ bf16_t f2bf(float f) {
  unsigned u = __builtin_bit_cast(unsigned, f);
  u += 0x7FFFu + ((u >> 16) & 1u);                // round-to-nearest-even
  unsigned short h = (unsigned short)(u >> 16);
  return __builtin_bit_cast(bf16_t, h);
}
__device__ __forceinline__ float bf2f(bf16_t b) {
  unsigned short h = __builtin_bit_cast(unsigned short, b);
  unsigned u = ((unsigned)h) << 16;
  return __builtin_bit_cast(float, u);
}
__device__ __forceinline__ float sigm(float x) { return 1.0f / (1.0f + __expf(-x)); }

__device__ __forceinline__ f32x8 fzero() {
  f32x8 z;
#pragma unroll
  for (int i = 0; i < 8; i++) z[i] = 0.0f;
  return z;
}

// A fragment: 16x32 bf16 tile, rows m=0..15 relative to `base`, row pitch `stride` (elems).
// ISA layout: lanes 0-15 hold M=lane, K runs {0..7,16..23}; lanes 16-31 K runs {8..15,24..31}.
__device__ __forceinline__ bf16x16 a_frag(const bf16_t* base, size_t stride) {
  int l = lane_id();
  const bf16_t* p = base + (size_t)(l & 15) * stride + (size_t)((l >> 4) << 3);
  bf16x8 lo = *(const bf16x8*)(p);
  bf16x8 hi = *(const bf16x8*)(p + 16);
  return __builtin_shufflevector(lo, hi, 0,1,2,3,4,5,6,7,8,9,10,11,12,13,14,15);
}

// B fragment: 32x16 bf16 tile from transposed weights Wt[N][K]; base -> (n0, k0), pitch `stride`.
// ISA layout: lanes 0-15 -> N=lane, K runs {0..7,16..23}; lanes 16-31 -> N=lane-16, K runs {8..15,24..31}.
__device__ __forceinline__ bf16x16 b_frag(const bf16_t* base, size_t stride) {
  int l = lane_id();
  const bf16_t* p = base + (size_t)(l & 15) * stride + (size_t)((l >> 4) << 3);
  bf16x8 lo = *(const bf16x8*)(p);
  bf16x8 hi = *(const bf16x8*)(p + 16);
  return __builtin_shufflevector(lo, hi, 0,1,2,3,4,5,6,7,8,9,10,11,12,13,14,15);
}

// B fragment with row clamp (for ragged N, e.g. fc4 N=1000). Lane-level address clamp only.
__device__ __forceinline__ bf16x16 b_frag_clamped(const bf16_t* mat, size_t stride,
                                                  int n0, int k0, int nmax) {
  int l = lane_id();
  int n = n0 + (l & 15); n = (n < nmax) ? n : (nmax - 1);
  const bf16_t* p = mat + (size_t)n * stride + (size_t)k0 + (size_t)((l >> 4) << 3);
  bf16x8 lo = *(const bf16x8*)(p);
  bf16x8 hi = *(const bf16x8*)(p + 16);
  return __builtin_shufflevector(lo, hi, 0,1,2,3,4,5,6,7,8,9,10,11,12,13,14,15);
}

__device__ __forceinline__ f32x8 wmma_bf(bf16x16 a, bf16x16 b, f32x8 c) {
  return __builtin_amdgcn_wmma_f32_16x16x32_bf16(false, a, false, b, (short)0, c, false, false);
}

#if USE_TDM
// Issue a TDM 2D tile load: `lines` rows of `tile0` bf16 elements, row stride
// `stride0` (elements), from global `gaddr` into LDS byte offset `lds_off`.
// Descriptor bit layout per CDNA5 ISA sec 8.3 (group0) / 8.4 (group1).
// clang-23 toolchain arity: (g0, g1, g2, g3, g4, cpol) = 6 args.
__device__ __forceinline__ void tdm_load_2d(unsigned lds_off, const void* gaddr,
                                            unsigned dim0, unsigned lines,
                                            unsigned tile0,
                                            unsigned long long stride0) {
  unsigned long long ga = (unsigned long long)(__SIZE_TYPE__)gaddr;
  u32x4 g0;
  g0[0] = 0x1u;                                        // count=1, user mode
  g0[1] = lds_off;                                     // lds_addr (bytes)
  g0[2] = (unsigned)(ga & 0xFFFFFFFFu);                // global_addr[31:0]
  g0[3] = (unsigned)((ga >> 32) & 0x1FFFFFFu) | (2u << 30);  // addr[56:32] | type=2
  i32x8 g1;
  g1[0] = (int)(1u << 16);                             // data_size=1 (2 bytes), mask=0
  g1[1] = (int)((dim0 & 0xFFFFu) << 16);               // tensor_dim0[15:0]
  g1[2] = (int)(((dim0 >> 16) & 0xFFFFu) | ((lines & 0xFFFFu) << 16));   // dim0 hi | dim1 lo
  g1[3] = (int)(((lines >> 16) & 0xFFFFu) | ((tile0 & 0xFFFFu) << 16));  // dim1 hi | tile_dim0
  g1[4] = (int)(lines & 0xFFFFu);                      // tile_dim1 = lines, tile_dim2 = 0
  g1[5] = (int)(unsigned)(stride0 & 0xFFFFFFFFull);    // tensor_dim0_stride[31:0]
  g1[6] = (int)(unsigned)((stride0 >> 32) & 0xFFFFull);// dim0_stride hi | dim1_stride lo = 0
  g1[7] = 0;
  i32x4 z4 = {0, 0, 0, 0};
  i32x8 z8 = {0, 0, 0, 0, 0, 0, 0, 0};
  __builtin_amdgcn_tensor_load_to_lds(g0, g1, z4, z4, z8, 0);
}
#endif

// ---------------------------------------------------------------------------
// Model constants
// ---------------------------------------------------------------------------
#define NB   32      // batch
#define TIN  2000
#define DDIN 80
#define NF   256     // conv filters
#define TE   999     // encoder length after stride-2 valid conv
#define UH   256     // encoder units per direction
#define H2   512     // decoder hidden
#define NV   1000    // vocab
#define TOUT 200
#define SOSI 1

// ---------------------------------------------------------------------------
// Conv1D (stride 2, valid) + bias + relu, emit bf16 [B][TE][NF]
// ---------------------------------------------------------------------------
__global__ __launch_bounds__(256) void k_conv(const float* __restrict__ x,
                                              const float* __restrict__ w,
                                              const float* __restrict__ bias,
                                              bf16_t* __restrict__ out) {
  int idx = blockIdx.x * 256 + threadIdx.x;
  if (idx >= NB * TE * NF) return;
  int f = idx & (NF - 1);
  int t = (idx >> 8) % TE;
  int b = idx / (TE * NF);
  const float* xp = x + ((size_t)b * TIN + 2 * t) * DDIN;
  float acc = bias[f];
#pragma unroll
  for (int k = 0; k < 3; k++)
    for (int d = 0; d < DDIN; d++)
      acc += xp[k * DDIN + d] * w[(k * DDIN + d) * NF + f];
  out[idx] = f2bf(fmaxf(acc, 0.0f));
}

// ---------------------------------------------------------------------------
// Generic transpose f32[K][N] -> bf16[N][K]
// ---------------------------------------------------------------------------
__global__ __launch_bounds__(256) void k_transpose(const float* __restrict__ in,
                                                   bf16_t* __restrict__ out,
                                                   int K, int N) {
  int idx = blockIdx.x * 256 + threadIdx.x;
  if (idx >= K * N) return;
  int k = idx / N, n = idx - k * N;
  out[(size_t)n * K + k] = f2bf(in[idx]);
}

// ---------------------------------------------------------------------------
// Encoder LSTM scan (one direction per block; grid.x = 2).
// 512 threads = 16 waves; wave w owns gate columns j in [16w,16w+16) and
// computes all 4 gates (two row-tile passes) via WMMA. h ping-pongs between
// two LDS buffers (read h_old / write h_new). K-loops are kept rolled
// (#pragma unroll 1): weights cannot be register-resident (512 KB/dir), so
// B-fragments stream from L2 each step; rolling the loop bounds live VGPRs
// and prevents scratch spills. The x_t tile (32 x Din, row stride Te*Din) is
// staged into LDS by the Tensor Data Mover, overlapped with the h-GEMM.
// ---------------------------------------------------------------------------
__global__ __launch_bounds__(512) void k_lstm_scan(
    const bf16_t* __restrict__ xin, int Din,
    const bf16_t* __restrict__ WtF, const bf16_t* __restrict__ UtF,
    const float* __restrict__ bF,
    const bf16_t* __restrict__ WtB, const bf16_t* __restrict__ UtB,
    const float* __restrict__ bB,
    bf16_t* __restrict__ out,
    float* __restrict__ hfinF, float* __restrict__ cfinF,
    float* __restrict__ hfinB, float* __restrict__ cfinB) {
  extern __shared__ char smem[];
  const int HP = UH + 8;                                  // padded pitch
  bf16_t* h_old = (bf16_t*)smem;                          // [32][HP]
  bf16_t* h_new = h_old + NB * HP;                        // [32][HP]
  float*  c_s   = (float*)(smem + 2 * NB * HP * 2);       // [32][UH]
  bf16_t* xtile = (bf16_t*)(smem + 2 * NB * HP * 2 + NB * UH * 4);  // [32][Din]
  const unsigned XTILE_OFF = (unsigned)(2 * NB * HP * 2 + NB * UH * 4);

  const int dir = blockIdx.x;
  const bf16_t* Wt = dir ? WtB : WtF;
  const bf16_t* Ut = dir ? UtB : UtF;
  const float*  bi = dir ? bB : bF;
  float* hfin = dir ? hfinB : hfinF;
  float* cfin = dir ? cfinB : cfinF;

  const int tid = threadIdx.x;
  const int lane = tid & 31, wave = tid >> 5;
  const int ln = lane & 15, lh = lane >> 4;
  const int j0 = wave * 16;

  for (int i = tid; i < 2 * NB * HP; i += blockDim.x) h_old[i] = f2bf(0.0f);
  for (int i = tid; i < NB * UH; i += blockDim.x) c_s[i] = 0.0f;
  __syncthreads();

  float bia[4];
#pragma unroll
  for (int g = 0; g < 4; g++) bia[g] = bi[g * UH + j0 + ln];

  const size_t xrow = (size_t)TE * Din;                   // batch stride in xin
  const int KX = Din >> 5;

  for (int s = 0; s < TE; s++) {
    const int ti = dir ? (TE - 1 - s) : s;
#if USE_TDM
    if (wave == 0)
      tdm_load_2d(XTILE_OFF, xin + (size_t)ti * Din, (unsigned)Din, NB,
                  (unsigned)Din, (unsigned long long)xrow);
#endif
    const int j = j0 + ln;
#pragma unroll
    for (int mt = 0; mt < 2; mt++) {
      f32x8 acc[4];
#pragma unroll
      for (int g = 0; g < 4; g++) acc[g] = fzero();

      // recurrent part: K = 256 over h_old (LDS); keep rolled (no spills)
      const bf16_t* hb = h_old + mt * 16 * HP;
#pragma unroll 1
      for (int ks = 0; ks < UH / 32; ks++) {
        bf16x16 a0 = a_frag(hb + ks * 32, HP);
#pragma unroll
        for (int g = 0; g < 4; g++) {
          bf16x16 bb = b_frag(Ut + (size_t)(g * UH + j0) * UH + ks * 32, UH);
          acc[g] = wmma_bf(a0, bb, acc[g]);
        }
      }
#if USE_TDM
      if (mt == 0) {                       // publish the TDM-staged x tile
        if (wave == 0) __builtin_amdgcn_s_wait_tensorcnt(0);
        __syncthreads();
      }
      const bf16_t* xb = xtile + (size_t)(mt * 16) * Din;
      const size_t xstride = (size_t)Din;
#else
      if (s + 1 < TE) {
        int tn = dir ? (TE - 2 - s) : (s + 1);
        if (lane == 0) __builtin_prefetch(xin + (size_t)tn * Din, 0, 0);
      }
      const bf16_t* xb = xin + (size_t)ti * Din + (size_t)(mt * 16) * xrow;
      const size_t xstride = xrow;
#endif
      // input part: K = Din over x_t; keep rolled
#pragma unroll 1
      for (int ks = 0; ks < KX; ks++) {
        bf16x16 a0 = a_frag(xb + ks * 32, xstride);
#pragma unroll
        for (int g = 0; g < 4; g++) {
          bf16x16 bb = b_frag(Wt + (size_t)(g * UH + j0) * Din + ks * 32, Din);
          acc[g] = wmma_bf(a0, bb, acc[g]);
        }
      }
      // cell update for rows mt*16 .. mt*16+15 (write h_new: no read hazard)
#pragma unroll
      for (int r = 0; r < 8; r++) {
        int m = mt * 16 + lh * 8 + r;
        float zi = acc[0][r] + bia[0];
        float zf = acc[1][r] + bia[1];
        float zg = acc[2][r] + bia[2];
        float zo = acc[3][r] + bia[3];
        float ig = sigm(zi), fg = sigm(zf), gg = tanhf(zg), og = sigm(zo);
        float cn = fg * c_s[m * UH + j] + ig * gg;
        c_s[m * UH + j] = cn;
        float hn = og * tanhf(cn);
        h_new[m * HP + j] = f2bf(hn);
        out[((size_t)m * TE + ti) * (2 * UH) + dir * UH + j] = f2bf(hn);
        if (s == TE - 1) { hfin[m * UH + j] = hn; cfin[m * UH + j] = cn; }
      }
    }
    __syncthreads();                        // publish h_new; xtile safe to reuse
    bf16_t* tmp = h_old; h_old = h_new; h_new = tmp;
  }
}

// ---------------------------------------------------------------------------
// Decoder: single persistent workgroup (1024 threads = 32 waves), 200 steps.
// Attention (VALU dots + wave32 shfl softmax) -> ctx; z-GEMM (WMMA, 4-gate
// wave tiling, two row-tile passes, h ping-pong) + one-hot token row-gather;
// LSTM cell; FC 512->512->256->64->1000 (WMMA); softmax -> out.
// ---------------------------------------------------------------------------
__global__ __launch_bounds__(1024) void k_decoder(
    const bf16_t* __restrict__ enc,          // [32][TE][512] bf16
    const float* __restrict__ hF, const float* __restrict__ cF,
    const float* __restrict__ hB, const float* __restrict__ cB,
    const int* __restrict__ y_true,          // [32][TOUT]
    const float* __restrict__ decW,          // [1512][2048] f32
    const float* __restrict__ decb,          // [2048]
    const bf16_t* __restrict__ Wct,          // [2048][512] (dec_W rows 1000..)
    const bf16_t* __restrict__ Utd,          // [2048][512]
    const bf16_t* __restrict__ fc1t, const float* __restrict__ fc1b,
    const bf16_t* __restrict__ fc2t, const float* __restrict__ fc2b,
    const bf16_t* __restrict__ fc3t, const float* __restrict__ fc3b,
    const bf16_t* __restrict__ fc4t, const float* __restrict__ fc4b,
    float* __restrict__ sc,                  // [32][TE]  scratch
    float* __restrict__ lg,                  // [32][1024] scratch
    float* __restrict__ outp) {              // [32][TOUT][1000]
  extern __shared__ char smem[];
  const int HP = H2 + 8;
  bf16_t* h_old = (bf16_t*)smem;                   // [32][HP]
  bf16_t* h_new = h_old + NB * HP;                 // [32][HP]
  bf16_t* ctx_s = h_new + NB * HP;                 // [32][HP]  (reused as fc2 out)
  bf16_t* f_s   = ctx_s + NB * HP;                 // [32][HP]  (fc1 out, fc3 out)
  float*  c_s   = (float*)(f_s + NB * HP);         // [32][H2]
  int*    tok_s = (int*)(c_s + NB * H2);           // [32]

  const int tid = threadIdx.x;
  const int lane = tid & 31, wave = tid >> 5;
  const int ln = lane & 15, lh = lane >> 4;
  const int j0 = wave * 16;

  // init decoder state: h = concat(hF,hB), c = concat(cF,cB)
  for (int i = tid; i < NB * H2; i += 1024) {
    int b = i >> 9, j = i & (H2 - 1);
    float hv = (j < UH) ? hF[b * UH + j] : hB[b * UH + j - UH];
    float cv = (j < UH) ? cF[b * UH + j] : cB[b * UH + j - UH];
    h_old[b * HP + j] = f2bf(hv);
    h_new[b * HP + j] = f2bf(0.0f);
    c_s[i] = cv;
  }
  __syncthreads();

  float db[4];
#pragma unroll
  for (int g = 0; g < 4; g++) db[g] = decb[g * H2 + j0 + ln];
  float fb1 = fc1b[j0 + ln];
  float fb2 = (wave < 16) ? fc2b[j0 + ln] : 0.0f;
  float fb3 = (wave < 4)  ? fc3b[j0 + ln] : 0.0f;
  float fb4a = fc4b[j0 + ln];
  int   n4b  = H2 + j0 + ln;
  float fb4b = (n4b < NV) ? fc4b[n4b] : 0.0f;

  for (int t = 0; t < TOUT; t++) {
    if (tid < NB) tok_s[tid] = (t == 0) ? SOSI : y_true[tid * TOUT + t - 1];
    __syncthreads();

    // ---- attention scores: sc[b][tp] = h[b] . enc[b][tp] ----
    for (int task = tid; task < NB * TE; task += 1024) {
      int b = task / TE, tp = task - b * TE;
      const bf16_t* e  = enc + ((size_t)b * TE + tp) * H2;
      const bf16_t* hh = h_old + b * HP;
      float acc = 0.0f;
      for (int k = 0; k < H2; k += 8) {
        bf16x8 ev = *(const bf16x8*)(e + k);
        bf16x8 hv = *(const bf16x8*)(hh + k);
#pragma unroll
        for (int q = 0; q < 8; q++) acc += bf2f(ev[q]) * bf2f(hv[q]);
      }
      sc[task] = acc;
    }
    __threadfence();
    __syncthreads();

    // ---- softmax over TE, wave w handles batch b=w ----
    {
      int b = wave;
      float mx = -3.4e38f;
      for (int tp = lane; tp < TE; tp += 32) mx = fmaxf(mx, sc[b * TE + tp]);
      for (int o = 16; o; o >>= 1) mx = fmaxf(mx, __shfl_xor(mx, o));
      float sum = 0.0f;
      for (int tp = lane; tp < TE; tp += 32) {
        float e = __expf(sc[b * TE + tp] - mx);
        sc[b * TE + tp] = e; sum += e;
      }
      for (int o = 16; o; o >>= 1) sum += __shfl_xor(sum, o);
      float inv = 1.0f / sum;
      for (int tp = lane; tp < TE; tp += 32) sc[b * TE + tp] *= inv;
    }
    __threadfence();
    __syncthreads();

    // ---- context: ctx[b][k] = sum_tp p[b][tp] * enc[b][tp][k] ----
    for (int task = tid; task < NB * H2; task += 1024) {
      int b = task >> 9, k = task & (H2 - 1);
      const bf16_t* e = enc + (size_t)b * TE * H2 + k;
      const float*  p = sc + b * TE;
      float acc = 0.0f;
      for (int tp = 0; tp < TE; tp++) acc += p[tp] * bf2f(e[(size_t)tp * H2]);
      ctx_s[b * HP + k] = f2bf(acc);
    }
    __syncthreads();

    // ---- z GEMM: ctx @ Wct^T + h @ Utd^T  (+ token row + bias); two mt passes
    const int j = j0 + ln;
#pragma unroll
    for (int mt = 0; mt < 2; mt++) {
      f32x8 acc[4];
#pragma unroll
      for (int g = 0; g < 4; g++) acc[g] = fzero();

      const bf16_t* cb = ctx_s + mt * 16 * HP;
#pragma unroll 1
      for (int ks = 0; ks < H2 / 32; ks++) {
        bf16x16 a0 = a_frag(cb + ks * 32, HP);
#pragma unroll
        for (int g = 0; g < 4; g++) {
          bf16x16 bb = b_frag(Wct + (size_t)(g * H2 + j0) * H2 + ks * 32, H2);
          acc[g] = wmma_bf(a0, bb, acc[g]);
        }
      }
      const bf16_t* hb = h_old + mt * 16 * HP;
#pragma unroll 1
      for (int ks = 0; ks < H2 / 32; ks++) {
        bf16x16 a0 = a_frag(hb + ks * 32, HP);
#pragma unroll
        for (int g = 0; g < 4; g++) {
          bf16x16 bb = b_frag(Utd + (size_t)(g * H2 + j0) * H2 + ks * 32, H2);
          acc[g] = wmma_bf(a0, bb, acc[g]);
        }
      }
      // cell update rows mt*16.. (writes h_new: no hazard with h_old readers)
#pragma unroll
      for (int r = 0; r < 8; r++) {
        int m = mt * 16 + lh * 8 + r;
        int tk = tok_s[m];
        float zi = acc[0][r] + db[0] + decW[(size_t)tk * 2048 + 0 * H2 + j];
        float zf = acc[1][r] + db[1] + decW[(size_t)tk * 2048 + 1 * H2 + j];
        float zg = acc[2][r] + db[2] + decW[(size_t)tk * 2048 + 2 * H2 + j];
        float zo = acc[3][r] + db[3] + decW[(size_t)tk * 2048 + 3 * H2 + j];
        float ig = sigm(zi), fg = sigm(zf), gg = tanhf(zg), og = sigm(zo);
        float cn = fg * c_s[m * H2 + j] + ig * gg;
        c_s[m * H2 + j] = cn;
        h_new[m * HP + j] = f2bf(og * tanhf(cn));
      }
    }
    __syncthreads();

    // ---- fc1: 512 -> 512 relu (all 32 waves), input h_new ----
    {
      f32x8 a_[2] = {fzero(), fzero()};
#pragma unroll 1
      for (int ks = 0; ks < H2 / 32; ks++) {
        bf16x16 a0 = a_frag(h_new + ks * 32, HP);
        bf16x16 a1 = a_frag(h_new + 16 * HP + ks * 32, HP);
        bf16x16 bb = b_frag(fc1t + (size_t)j0 * H2 + ks * 32, H2);
        a_[0] = wmma_bf(a0, bb, a_[0]);
        a_[1] = wmma_bf(a1, bb, a_[1]);
      }
#pragma unroll
      for (int mt = 0; mt < 2; mt++)
#pragma unroll
        for (int r = 0; r < 8; r++) {
          int m = mt * 16 + lh * 8 + r;
          f_s[m * HP + j] = f2bf(fmaxf(a_[mt][r] + fb1, 0.0f));
        }
    }
    __syncthreads();

    // ---- fc2: 512 -> 256 relu (waves 0..15) ----
    if (wave < 16) {
      f32x8 a_[2] = {fzero(), fzero()};
#pragma unroll 1
      for (int ks = 0; ks < H2 / 32; ks++) {
        bf16x16 a0 = a_frag(f_s + ks * 32, HP);
        bf16x16 a1 = a_frag(f_s + 16 * HP + ks * 32, HP);
        bf16x16 bb = b_frag(fc2t + (size_t)j0 * H2 + ks * 32, H2);
        a_[0] = wmma_bf(a0, bb, a_[0]);
        a_[1] = wmma_bf(a1, bb, a_[1]);
      }
#pragma unroll
      for (int mt = 0; mt < 2; mt++)
#pragma unroll
        for (int r = 0; r < 8; r++) {
          int m = mt * 16 + lh * 8 + r;
          ctx_s[m * HP + j] = f2bf(fmaxf(a_[mt][r] + fb2, 0.0f));
        }
    }
    __syncthreads();

    // ---- fc3: 256 -> 64 relu (waves 0..3) ----
    if (wave < 4) {
      f32x8 a_[2] = {fzero(), fzero()};
#pragma unroll 1
      for (int ks = 0; ks < 256 / 32; ks++) {
        bf16x16 a0 = a_frag(ctx_s + ks * 32, HP);
        bf16x16 a1 = a_frag(ctx_s + 16 * HP + ks * 32, HP);
        bf16x16 bb = b_frag(fc3t + (size_t)j0 * 256 + ks * 32, 256);
        a_[0] = wmma_bf(a0, bb, a_[0]);
        a_[1] = wmma_bf(a1, bb, a_[1]);
      }
#pragma unroll
      for (int mt = 0; mt < 2; mt++)
#pragma unroll
        for (int r = 0; r < 8; r++) {
          int m = mt * 16 + lh * 8 + r;
          f_s[m * HP + j] = f2bf(fmaxf(a_[mt][r] + fb3, 0.0f));
        }
    }
    __syncthreads();

    // ---- fc4: 64 -> 1000 logits (2 n-windows per wave) ----
    {
      f32x8 a_[2][2] = {{fzero(), fzero()}, {fzero(), fzero()}};
      const bool win2 = (H2 + j0) < NV;   // wave-uniform
      for (int ks = 0; ks < 2; ks++) {
        bf16x16 a0 = a_frag(f_s + ks * 32, HP);
        bf16x16 a1 = a_frag(f_s + 16 * HP + ks * 32, HP);
        bf16x16 b0 = b_frag_clamped(fc4t, 64, j0, ks * 32, NV);
        a_[0][0] = wmma_bf(a0, b0, a_[0][0]);
        a_[0][1] = wmma_bf(a1, b0, a_[0][1]);
        if (win2) {
          bf16x16 b1 = b_frag_clamped(fc4t, 64, H2 + j0, ks * 32, NV);
          a_[1][0] = wmma_bf(a0, b1, a_[1][0]);
          a_[1][1] = wmma_bf(a1, b1, a_[1][1]);
        }
      }
      const int ja = j0 + ln;
      const int jb = H2 + j0 + ln;
#pragma unroll
      for (int mt = 0; mt < 2; mt++)
#pragma unroll
        for (int r = 0; r < 8; r++) {
          int m = mt * 16 + lh * 8 + r;
          lg[m * 1024 + ja] = a_[0][mt][r] + fb4a;
          if (jb < NV) lg[m * 1024 + jb] = a_[1][mt][r] + fb4b;
        }
    }
    __threadfence();
    __syncthreads();

    // ---- output softmax over vocab, wave w handles batch b=w ----
    {
      int b = wave;
      float mx = -3.4e38f;
      for (int v = lane; v < NV; v += 32) mx = fmaxf(mx, lg[b * 1024 + v]);
      for (int o = 16; o; o >>= 1) mx = fmaxf(mx, __shfl_xor(mx, o));
      float sum = 0.0f;
      for (int v = lane; v < NV; v += 32) {
        float e = __expf(lg[b * 1024 + v] - mx);
        lg[b * 1024 + v] = e; sum += e;
      }
      for (int o = 16; o; o >>= 1) sum += __shfl_xor(sum, o);
      float inv = 1.0f / sum;
      float* op = outp + ((size_t)b * TOUT + t) * NV;
      for (int v = lane; v < NV; v += 32) op[v] = lg[b * 1024 + v] * inv;
    }
    __syncthreads();
    bf16_t* tmp = h_old; h_old = h_new; h_new = tmp;
  }
}

// ---------------------------------------------------------------------------
// Host orchestration
// ---------------------------------------------------------------------------
extern "C" void kernel_launch(void* const* d_in, const int* in_sizes, int n_in,
                              void* d_out, int out_size, void* d_ws, size_t ws_size,
                              hipStream_t stream) {
  (void)in_sizes; (void)n_in; (void)out_size; (void)ws_size;
  const float* x       = (const float*)d_in[0];
  const int*   y_true  = (const int*)d_in[1];
  const float* conv_w  = (const float*)d_in[2];
  const float* conv_b  = (const float*)d_in[3];
  const float* e0fW = (const float*)d_in[4],  *e0fU = (const float*)d_in[5],  *e0fb = (const float*)d_in[6];
  const float* e0bW = (const float*)d_in[7],  *e0bU = (const float*)d_in[8],  *e0bb = (const float*)d_in[9];
  const float* e1fW = (const float*)d_in[10], *e1fU = (const float*)d_in[11], *e1fb = (const float*)d_in[12];
  const float* e1bW = (const float*)d_in[13], *e1bU = (const float*)d_in[14], *e1bb = (const float*)d_in[15];
  const float* decW = (const float*)d_in[16], *decU = (const float*)d_in[17], *decb = (const float*)d_in[18];
  const float* fc1w = (const float*)d_in[19], *fc1b = (const float*)d_in[20];
  const float* fc2w = (const float*)d_in[21], *fc2b = (const float*)d_in[22];
  const float* fc3w = (const float*)d_in[23], *fc3b = (const float*)d_in[24];
  const float* fc4w = (const float*)d_in[25], *fc4b = (const float*)d_in[26];

  char* base = (char*)d_ws;
  size_t off = 0;
  auto bump = [&](size_t bytes) -> char* {
    char* p = base + off;
    off += (bytes + 255) & ~(size_t)255;
    return p;
  };

  bf16_t* conv_out = (bf16_t*)bump((size_t)NB * TE * NF * 2);
  bf16_t* enc0_out = (bf16_t*)bump((size_t)NB * TE * 2 * UH * 2);
  bf16_t* enc1_out = (bf16_t*)bump((size_t)NB * TE * 2 * UH * 2);
  bf16_t* w0f_t = (bf16_t*)bump((size_t)1024 * 256 * 2);
  bf16_t* u0f_t = (bf16_t*)bump((size_t)1024 * 256 * 2);
  bf16_t* w0b_t = (bf16_t*)bump((size_t)1024 * 256 * 2);
  bf16_t* u0b_t = (bf16_t*)bump((size_t)1024 * 256 * 2);
  bf16_t* w1f_t = (bf16_t*)bump((size_t)1024 * 512 * 2);
  bf16_t* u1f_t = (bf16_t*)bump((size_t)1024 * 256 * 2);
  bf16_t* w1b_t = (bf16_t*)bump((size_t)1024 * 512 * 2);
  bf16_t* u1b_t = (bf16_t*)bump((size_t)1024 * 256 * 2);
  bf16_t* wct_t = (bf16_t*)bump((size_t)2048 * 512 * 2);
  bf16_t* utd_t = (bf16_t*)bump((size_t)2048 * 512 * 2);
  bf16_t* fc1_t = (bf16_t*)bump((size_t)512 * 512 * 2);
  bf16_t* fc2_t = (bf16_t*)bump((size_t)256 * 512 * 2);
  bf16_t* fc3_t = (bf16_t*)bump((size_t)64 * 256 * 2);
  bf16_t* fc4_t = (bf16_t*)bump((size_t)1000 * 64 * 2);
  float* hf_f = (float*)bump((size_t)NB * UH * 4);
  float* cf_f = (float*)bump((size_t)NB * UH * 4);
  float* hf_b = (float*)bump((size_t)NB * UH * 4);
  float* cf_b = (float*)bump((size_t)NB * UH * 4);
  float* sc   = (float*)bump((size_t)NB * TE * 4);
  float* lg   = (float*)bump((size_t)NB * 1024 * 4);

  auto tr = [&](const float* in, bf16_t* out, int K, int N) {
    int n = K * N;
    k_transpose<<<dim3((n + 255) / 256), dim3(256), 0, stream>>>(in, out, K, N);
  };
  tr(e0fW, w0f_t, 256, 1024); tr(e0fU, u0f_t, 256, 1024);
  tr(e0bW, w0b_t, 256, 1024); tr(e0bU, u0b_t, 256, 1024);
  tr(e1fW, w1f_t, 512, 1024); tr(e1fU, u1f_t, 256, 1024);
  tr(e1bW, w1b_t, 512, 1024); tr(e1bU, u1b_t, 256, 1024);
  tr(decW + (size_t)NV * 2048, wct_t, 512, 2048);   // ctx part of dec_W
  tr(decU, utd_t, 512, 2048);
  tr(fc1w, fc1_t, 512, 512);
  tr(fc2w, fc2_t, 512, 256);
  tr(fc3w, fc3_t, 256, 64);
  tr(fc4w, fc4_t, 64, 1000);

  {
    int n = NB * TE * NF;
    k_conv<<<dim3((n + 255) / 256), dim3(256), 0, stream>>>(x, conv_w, conv_b, conv_out);
  }

  auto scan_smem = [](int Din) -> size_t {
    return (size_t)2 * NB * (UH + 8) * 2 + (size_t)NB * UH * 4 + (size_t)NB * Din * 2;
  };
  k_lstm_scan<<<dim3(2), dim3(512), scan_smem(NF), stream>>>(
      conv_out, NF, w0f_t, u0f_t, e0fb, w0b_t, u0b_t, e0bb,
      enc0_out, hf_f, cf_f, hf_b, cf_b);
  k_lstm_scan<<<dim3(2), dim3(512), scan_smem(2 * UH), stream>>>(
      enc0_out, 2 * UH, w1f_t, u1f_t, e1fb, w1b_t, u1b_t, e1bb,
      enc1_out, hf_f, cf_f, hf_b, cf_b);

  const size_t dec_smem = (size_t)4 * NB * (H2 + 8) * 2 + (size_t)NB * H2 * 4 + 256;
  k_decoder<<<dim3(1), dim3(1024), dec_smem, stream>>>(
      enc1_out, hf_f, cf_f, hf_b, cf_b, y_true, decW, decb,
      wct_t, utd_t, fc1_t, fc1b, fc2_t, fc2b, fc3_t, fc3b, fc4_t, fc4b,
      sc, lg, (float*)d_out);
}
